// BiQRNN_8186207666212
// MI455X (gfx1250) — matrix operand order repaired and checked by reading, not compiled, not run on hardware
//
#include <hip/hip_runtime.h>
#include <hip/hip_bf16.h>
#include <math.h>
#include <stdint.h>

typedef _Float16 f16;
typedef __attribute__((ext_vector_type(16))) _Float16 v16h;
typedef __attribute__((ext_vector_type(8)))  _Float16 v8h;
typedef __attribute__((ext_vector_type(8)))  float    v8f;

#define EMB     256
#define HID     512
#define NUM_IN  7
#define NUM_OUT 4
#define Bb      64
#define Ss      512
#define Dp      288   // D=260 zero-padded to 9*32 for the WMMA K loop
#define NCHUNK  32    // parallel-scan chunks over S
#define SCHUNK  (Ss / NCHUNK)   // 16 timesteps per chunk
#define ROWS    32    // batch rows per block (2 waves)

union V16 { v16h v; v8h h[2]; };

__device__ __forceinline__ float fast_sigmoid(float x) {
    return __builtin_amdgcn_rcpf(1.f + __expf(-x));            // branchless: v_exp + v_rcp
}
__device__ __forceinline__ float fast_tanh(float x) {
    return 2.f * __builtin_amdgcn_rcpf(1.f + __expf(-2.f * x)) - 1.f;
}

// CDNA5 async copy: 16B per lane, global -> LDS, ASYNCcnt-tracked, no VGPR dest.
// INST_OFFSET is applied to BOTH global and LDS addresses (ISA §15.18.3), so the
// per-chunk advance lives in the immediate and the bases stay in one VGPR (pair).
template <int IMM>
__device__ __forceinline__ void async_copy16(uint32_t lds_addr, const void* gaddr) {
    asm volatile("global_load_async_to_lds_b128 %0, %1, off offset:%2"
                 :: "v"(lds_addr), "v"(gaddr), "n"(IMM) : "memory");
}

template <int J>
struct AsyncBatch {
    static __device__ __forceinline__ void run(uint32_t lds, const void* g) {
        AsyncBatch<J - 1>::run(lds, g);
        async_copy16<(J - 1) * 512>(lds, g);
    }
};
template <> struct AsyncBatch<0> {
    static __device__ __forceinline__ void run(uint32_t, const void*) {}
};

// ---------- K0a: Wf gate columns (first 2*HID of 1536) -> f16, transposed [n][k], K padded
__global__ void prep_wT(const float* __restrict__ W, f16* __restrict__ WT) {
    int idx = blockIdx.x * blockDim.x + threadIdx.x;
    if (idx >= 1024 * Dp) return;
    int n = idx / Dp, k = idx % Dp;
    float v = (k < 260) ? W[(size_t)k * 1536 + n] : 0.f;
    WT[(size_t)n * Dp + k] = (f16)v;
}

// ---------- K0b: padded f16 activations, s-major: c16[s][b][Dp] = [emb[ev], num@Wn+bn, 0...]
__global__ void prep_c(const float* __restrict__ X, const float* __restrict__ emb,
                       const float* __restrict__ Wn, const float* __restrict__ bn,
                       f16* __restrict__ c16) {
    int bs = blockIdx.x;               // b*S + s  (input is [b][s][8])
    int b  = bs / Ss, s = bs % Ss;
    int d  = threadIdx.x;              // 0..Dp-1 (blockDim = 288 = 9 waves)
    const float* x = X + (size_t)bs * 8;
    float v = 0.f;
    if (d < EMB) {
        int ev = (int)x[0];
        v = emb[(size_t)ev * EMB + d];
    } else if (d < EMB + NUM_OUT) {
        int q = d - EMB;
        float acc = bn[q];
        #pragma unroll
        for (int i = 0; i < NUM_IN; ++i) acc += x[1 + i] * Wn[i * NUM_OUT + q];
        v = acc;
    }
    c16[((size_t)s * Bb + b) * Dp + d] = (f16)v;     // s-major: wave tiles are contiguous
}

// ---------- K1: chunked forward gates (WMMA f16) + local affine scan composition.
// Async-LDS double-buffered pipeline: prefetch step s+1 while WMMAs consume step s from LDS.
__global__ __launch_bounds__(64) void fwd_scan(
    const f16* __restrict__ c16, const f16* __restrict__ WfT,
    const float* __restrict__ bfv, float* __restrict__ Ac, float* __restrict__ Bc) {
    __shared__ __align__(16) f16 SL[2][ROWS][Dp];       // 36 KB double buffer
    const int chunk = blockIdx.x;            // 0..NCHUNK-1 (s chunk)
    const int n0    = blockIdx.y * 16;       // hidden-column tile
    const int bh    = blockIdx.z;            // batch half (0/1)
    const int tid   = threadIdx.x;
    const int wave  = tid >> 5;              // 0..1
    const int lane  = tid & 31;
    const int m0    = wave * 16;             // block-local batch tile
    const int nl    = lane & 15;
    const int hi    = lane >> 4;
    const int lo8   = hi * 8;                // A-frag K sub-offset
    const int lo16  = hi * 8;                // B-frag K sub-offset

    // B fragments (weights) straight from global into registers for the whole chunk
    V16 bz[Dp / 32], bg[Dp / 32];
    #pragma unroll
    for (int kc = 0; kc < Dp / 32; ++kc) {
        const f16* wz = WfT + (size_t)(n0 + nl) * Dp + kc * 32 + lo16;
        const f16* wf = WfT + (size_t)(HID + n0 + nl) * Dp + kc * 32 + lo16;
        bz[kc].h[0] = *(const v8h*)wz;  bz[kc].h[1] = *(const v8h*)(wz + 16);
        bg[kc].h[0] = *(const v8h*)wf;  bg[kc].h[1] = *(const v8h*)(wf + 16);
    }
    const float zbias = bfv[n0 + nl];
    const float fbias = bfv[HID + n0 + nl];

    // Wave-private async staging: each wave copies its own contiguous 9216B row-slab.
    // Generic-pointer low 32 bits == LDS byte offset (ISA aperture rule).
    const uint32_t lds0 = (uint32_t)(uintptr_t)&SL[0][0][0] + (uint32_t)(m0 * Dp * 2) + lane * 16;
    const uint32_t bufstride = ROWS * Dp * 2;
    const char* gsrc0 = (const char*)c16 + ((size_t)(bh * ROWS + m0)) * Dp * 2 + lane * 16;
    const int s0 = chunk * SCHUNK;

    float areg[8], breg[8];                  // local affine composition (A,B)
    #pragma unroll
    for (int r = 0; r < 8; ++r) { areg[r] = 1.f; breg[r] = 0.f; }

    // 18 async copies move one 9216B wave-slab; offsets are immediates, bases are loop-invariant
    AsyncBatch<18>::run(lds0, gsrc0 + (size_t)s0 * (Bb * Dp * 2));
    int buf = 0;
    for (int i = 0; i < SCHUNK; ++i) {
        if (i + 1 < SCHUNK) {
            AsyncBatch<18>::run(lds0 + (buf ^ 1) * bufstride,
                                gsrc0 + (size_t)(s0 + i + 1) * (Bb * Dp * 2));
            asm volatile("s_wait_asynccnt 18" ::: "memory");   // batch i landed (in-order)
        } else {
            asm volatile("s_wait_asynccnt 0" ::: "memory");
        }
        v8f accZ, accF;                      // fold bias into accumulator init
        #pragma unroll
        for (int r = 0; r < 8; ++r) { accZ[r] = zbias; accF[r] = fbias; }
        #pragma unroll
        for (int kc = 0; kc < Dp / 32; ++kc) {
            V16 a;                           // A fragment from LDS (ds_load_b128 x2)
            a.h[0] = *(const v8h*)&SL[buf][m0 + nl][kc * 32 + lo8];
            a.h[1] = *(const v8h*)&SL[buf][m0 + nl][kc * 32 + 16 + lo8];
            accZ = __builtin_amdgcn_wmma_f32_16x16x32_f16(false, a.v, false, bz[kc].v,
                                                          (short)0, accZ, false, false);
            accF = __builtin_amdgcn_wmma_f32_16x16x32_f16(false, a.v, false, bg[kc].v,
                                                          (short)0, accF, false, false);
        }
        #pragma unroll
        for (int r = 0; r < 8; ++r) {        // branchless TRANS ops co-execute with WMMA
            float z = fast_tanh(accZ[r]);
            float f = fast_sigmoid(accF[r]);
            areg[r] *= f;
            breg[r] = f * breg[r] + (1.f - f) * z;
        }
        buf ^= 1;
    }
    #pragma unroll
    for (int r = 0; r < 8; ++r) {
        int brow = bh * ROWS + m0 + r + 8 * hi;    // C/D layout: M = r + 8*(lane/16)
        size_t o = (size_t)chunk * Bb * HID + (size_t)brow * HID + n0 + nl;
        Ac[o] = areg[r];
        Bc[o] = breg[r];
    }
}

// ---------- K1b: combine the 32 chunk-affine pairs -> hf[S-1]
__global__ void scan_combine(const float* __restrict__ Ac, const float* __restrict__ Bc,
                             float* __restrict__ hf_last) {
    int idx = blockIdx.x * blockDim.x + threadIdx.x;
    if (idx >= Bb * HID) return;
    float h = 0.f;
    #pragma unroll
    for (int c = 0; c < NCHUNK; ++c) {
        size_t o = (size_t)c * Bb * HID + idx;
        h = Ac[o] * h + Bc[o];
    }
    hf_last[idx] = h;
}

// ---------- K2: backward branch collapses to one timestep: hb[-1] = (1-sig(F))*tanh(Z) at s=S-1
__global__ void bwd_last(const f16* __restrict__ c16, const float* __restrict__ Wb,
                         const float* __restrict__ bb, float* __restrict__ hb_last) {
    int idx = blockIdx.x * blockDim.x + threadIdx.x;
    if (idx >= Bb * HID) return;
    int b = idx / HID, j = idx % HID;
    const f16* cp = c16 + ((size_t)(Ss - 1) * Bb + b) * Dp;
    float gz = bb[j], gf = bb[HID + j];
    for (int d = 0; d < 260; ++d) {
        float cd = (float)cp[d];
        gz += cd * Wb[(size_t)d * 1536 + j];
        gf += cd * Wb[(size_t)d * 1536 + HID + j];
    }
    float f = fast_sigmoid(gf);
    hb_last[idx] = (1.f - f) * fast_tanh(gz);
}

// ---------- K3: out[b] = [hf_last, hb_last] . Wo + bo
__global__ void out_proj(const float* __restrict__ hf, const float* __restrict__ hb,
                         const float* __restrict__ Wo, const float* __restrict__ bo,
                         float* __restrict__ out) {
    __shared__ float red[256];
    int b = blockIdx.x, t = threadIdx.x;
    float s = 0.f;
    for (int k = t; k < 2 * HID; k += 256) {
        float v = (k < HID) ? hf[(size_t)b * HID + k] : hb[(size_t)b * HID + k - HID];
        s += v * Wo[k];
    }
    red[t] = s;
    __syncthreads();
    for (int off = 128; off > 0; off >>= 1) {
        if (t < off) red[t] += red[t + off];
        __syncthreads();
    }
    if (t == 0) out[b] = red[0] + bo[0];
}

extern "C" void kernel_launch(void* const* d_in, const int* in_sizes, int n_in,
                              void* d_out, int out_size, void* d_ws, size_t ws_size,
                              hipStream_t stream) {
    const float* X   = (const float*)d_in[0];
    const float* emb = (const float*)d_in[1];
    const float* Wn  = (const float*)d_in[2];
    const float* bn  = (const float*)d_in[3];
    const float* Wf  = (const float*)d_in[4];
    const float* bfv = (const float*)d_in[5];
    const float* Wb  = (const float*)d_in[6];
    const float* bb  = (const float*)d_in[7];
    const float* Wo  = (const float*)d_in[8];
    const float* bo  = (const float*)d_in[9];
    float* out = (float*)d_out;

    char* ws = (char*)d_ws;
    f16* c16 = (f16*)ws;                 size_t off = (size_t)Bb * Ss * Dp * sizeof(f16);
    f16* WfT = (f16*)(ws + off);         off += (size_t)1024 * Dp * sizeof(f16);
    float* Ac = (float*)(ws + off);      off += (size_t)NCHUNK * Bb * HID * sizeof(float);
    float* Bc = (float*)(ws + off);      off += (size_t)NCHUNK * Bb * HID * sizeof(float);
    float* hf_last = (float*)(ws + off); off += (size_t)Bb * HID * sizeof(float);
    float* hb_last = (float*)(ws + off);

    prep_wT <<<(1024 * Dp + 255) / 256, 256, 0, stream>>>(Wf, WfT);
    prep_c  <<<Bb * Ss, Dp, 0, stream>>>(X, emb, Wn, bn, c16);
    fwd_scan<<<dim3(NCHUNK, HID / 16, 2), 64, 0, stream>>>(c16, WfT, bfv, Ac, Bc);
    scan_combine<<<(Bb * HID + 255) / 256, 256, 0, stream>>>(Ac, Bc, hf_last);
    bwd_last<<<(Bb * HID + 255) / 256, 256, 0, stream>>>(c16, Wb, bb, hb_last);
    out_proj<<<Bb, 256, 0, stream>>>(hf_last, hb_last, Wo, bo, out);
}
